// MultiHeadLatentAttentionCustom_33526514712927
// MI455X (gfx1250) — matrix operand order repaired and checked by reading, not compiled, not run on hardware
//
#include <hip/hip_runtime.h>
#include <hip/hip_bf16.h>
#include <math.h>

typedef _Float16 f16;
typedef __attribute__((ext_vector_type(16))) _Float16 v16h;
typedef __attribute__((ext_vector_type(8)))  _Float16 v8h;
typedef __attribute__((ext_vector_type(8)))  float    v8f;

#define SHUF16(lo, hi) __builtin_shufflevector((lo), (hi), \
    0,1,2,3,4,5,6,7,8,9,10,11,12,13,14,15)

// ---- CDNA5 async global->LDS copies (ASYNCcnt), with safe fallback --------
#if __has_builtin(__builtin_amdgcn_global_load_async_to_lds_b128)
#define GEMM_ASYNC 1
#else
#define GEMM_ASYNC 0
#endif

#if GEMM_ASYNC
// Signature (from hipcc diagnostic): param0 = global (AS1) 'int __vector(4)*'.
typedef int v4i_g __attribute__((vector_size(16)));
typedef __attribute__((address_space(1))) v4i_g* as1_v4ip;
typedef __attribute__((address_space(3))) v4i_g* as3_v4ip;

__device__ __forceinline__ void async_cp16(const f16* g, f16* l) {
  __builtin_amdgcn_global_load_async_to_lds_b128(
      (as1_v4ip)(unsigned long long)(const void*)g,
      (as3_v4ip)(unsigned int)(unsigned long long)(const void*)l,
      /*offset=*/0, /*cpol=*/0);
}
#endif

__device__ __forceinline__ void wait_async0() {
#if __has_builtin(__builtin_amdgcn_s_wait_asynccnt)
  __builtin_amdgcn_s_wait_asynccnt(0);
#else
  asm volatile("s_wait_asynccnt 0x0" ::: "memory");
#endif
}

// ---------------------------------------------------------------- f32 -> f16
__global__ __launch_bounds__(256)
void cvt_f32_to_f16(const float* __restrict__ in, f16* __restrict__ out, int n) {
  int i = blockIdx.x * 256 + threadIdx.x;
  if (i < n) out[i] = (f16)in[i];
}

// ------------------------------------------------- transpose (+convert) to f16
// in: [R][C] (SRC_T), out: [C][R] (f16).  32x32 LDS tile, 256 threads.
template<typename SRC_T>
__global__ __launch_bounds__(256)
void transpose_to_f16(const SRC_T* __restrict__ in, f16* __restrict__ out,
                      int R, int C) {
  __shared__ f16 tile[32][33];
  const int tx = threadIdx.x & 31;
  const int ty = threadIdx.x >> 5;       // 0..7
  const int c0 = blockIdx.x * 32;
  const int r0 = blockIdx.y * 32;
  #pragma unroll
  for (int i = 0; i < 32; i += 8) {
    const int r = r0 + ty + i, c = c0 + tx;
    tile[ty + i][tx] = (r < R && c < C) ? (f16)in[(size_t)r * C + c] : (f16)0.f;
  }
  __syncthreads();
  #pragma unroll
  for (int i = 0; i < 32; i += 8) {
    const int c = c0 + ty + i, r = r0 + tx;
    if (c < C && r < R) out[(size_t)c * R + r] = tile[tx][ty + i];
  }
}

// ---------------------------------------------------------------- WMMA GEMM
// C[M,N] = alpha * (A[M,K] @ Bt^T[K,N]) + bias[N]
// A row-major f16 (lda).  Bt is B transposed: [N][K] row-major (ldb).
// Block tile 128x128x32, 8 wave32 waves, wave patch 64x32 = 4x2 wmma tiles.
// Double-buffered LDS; both tiles staged with async global->LDS b128.
// Callers guarantee M % 128 == 0 and K % 32 == 0; N may be partial
// (staging rows clamped to N-1; stores guarded).
template<bool OUT_F16>
__global__ __launch_bounds__(256)
void gemm_wmma_f16(const f16* __restrict__ A, int lda,
                   const f16* __restrict__ Bt, int ldb,
                   void* __restrict__ Cv, int ldc,
                   const float* __restrict__ bias,
                   int M, int N, int K, float alpha)
{
  __shared__ __align__(16) f16 As[2][128 * 32];   // [m][k]
  __shared__ __align__(16) f16 Bs[2][128 * 32];   // [n][k]

  const int tid  = threadIdx.x;
  const int lane = tid & 31;
  const int wave = tid >> 5;
  const int half = lane >> 4;    // 0 | 1
  const int l16  = lane & 15;
  const int wm   = wave >> 2;    // 0..1  -> 64-row strip
  const int wn   = wave & 3;     // 0..3  -> 32-col strip
  const int rowBase = blockIdx.y * 128;
  const int colBase = blockIdx.x * 128;

  v8f acc[4][2];
  #pragma unroll
  for (int i = 0; i < 4; ++i)
    #pragma unroll
    for (int j = 0; j < 2; ++j)
      #pragma unroll
      for (int e = 0; e < 8; ++e) acc[i][j][e] = 0.0f;

  const int ar = tid >> 1;            // 0..127 (staging row)
  const int ac = (tid & 1) << 4;      // 0 | 16 (f16 units)
  const int bn = min(colBase + ar, N - 1);   // clamp: partial-N safe reads

  // Stage one 128x32 K-slice into LDS buffer `buf`.
  auto stage = [&](int buf, int k0) {
    const f16* srcA = A  + (size_t)(rowBase + ar) * lda + k0 + ac;
    const f16* srcB = Bt + (size_t)bn * ldb + k0 + ac;
    f16* dstA = &As[buf][ar * 32 + ac];
    f16* dstB = &Bs[buf][ar * 32 + ac];
#if GEMM_ASYNC
    async_cp16(srcA,     dstA);
    async_cp16(srcA + 8, dstA + 8);
    async_cp16(srcB,     dstB);
    async_cp16(srcB + 8, dstB + 8);
#else
    uint4 a0 = *(const uint4*)(srcA);
    uint4 a1 = *(const uint4*)(srcA + 8);
    uint4 b0 = *(const uint4*)(srcB);
    uint4 b1 = *(const uint4*)(srcB + 8);
    *(uint4*)(dstA)     = a0;
    *(uint4*)(dstA + 8) = a1;
    *(uint4*)(dstB)     = b0;
    *(uint4*)(dstB + 8) = b1;
#endif
    __builtin_prefetch(srcA + 64, 0, 1);   // global_prefetch_b8, tile after next
  };

  // ---- software pipeline: prologue stage, then compute(i) || stage(i+1)
  stage(0, 0);
#if GEMM_ASYNC
  wait_async0();
#endif
  __syncthreads();

  int cur = 0;
  for (int k0 = 0; k0 < K; k0 += 32) {
    const int k1 = k0 + 32;
    if (k1 < K) stage(cur ^ 1, k1);     // async fill of the other buffer

    // ---- fragments (contiguous k = 16*half + s, same in A and B) + 8 WMMAs from buffer `cur`
    v16h bf[2];
    #pragma unroll
    for (int nt = 0; nt < 2; ++nt) {
      const int c = wn * 32 + nt * 16 + l16;
      v8h lo = *(const v8h*)&Bs[cur][c * 32 + half * 16];
      v8h hi = *(const v8h*)&Bs[cur][c * 32 + half * 16 + 8];
      bf[nt] = SHUF16(lo, hi);
    }
    #pragma unroll
    for (int mt = 0; mt < 4; ++mt) {
      const int r = wm * 64 + mt * 16 + l16;
      v8h lo = *(const v8h*)&As[cur][r * 32 + half * 16];
      v8h hi = *(const v8h*)&As[cur][r * 32 + half * 16 + 8];
      v16h af = SHUF16(lo, hi);
      acc[mt][0] = __builtin_amdgcn_wmma_f32_16x16x32_f16(
          false, af, false, bf[0], (short)0, acc[mt][0], false, false);
      acc[mt][1] = __builtin_amdgcn_wmma_f32_16x16x32_f16(
          false, af, false, bf[1], (short)0, acc[mt][1], false, false);
    }

    if (k1 < K) {
#if GEMM_ASYNC
      wait_async0();                    // my async stores to cur^1 landed
#endif
      __syncthreads();                  // + dscnt wait: everyone done with cur
      cur ^= 1;
    }
  }

  // ---- epilogue: D layout M = vgpr + 8*half, N = lane%16
  #pragma unroll
  for (int mt = 0; mt < 4; ++mt)
    #pragma unroll
    for (int nt = 0; nt < 2; ++nt)
      #pragma unroll
      for (int r = 0; r < 8; ++r) {
        const int row = rowBase + wm * 64 + mt * 16 + r + half * 8;
        const int col = colBase + wn * 32 + nt * 16 + l16;
        if (row < M && col < N) {
          float v = acc[mt][nt][r] * alpha;
          if (bias) v += bias[col];
          if (OUT_F16) ((f16*)Cv)[(size_t)row * ldc + col] = (f16)v;
          else         ((float*)Cv)[(size_t)row * ldc + col] = v;
        }
      }
}

// ---------------------------------------------------------------- softmax rows
__global__ __launch_bounds__(256)
void softmax_rows(const float* __restrict__ S, f16* __restrict__ P, int n) {
  __shared__ float red[256];
  const int row = blockIdx.x;
  const int tid = threadIdx.x;
  const float* s = S + (size_t)row * n;
  f16* p = P + (size_t)row * n;

  float m = -3.402823466e38f;
  for (int i = tid; i < n; i += 256) m = fmaxf(m, s[i]);
  red[tid] = m; __syncthreads();
  for (int off = 128; off > 0; off >>= 1) {
    if (tid < off) red[tid] = fmaxf(red[tid], red[tid + off]);
    __syncthreads();
  }
  m = red[0]; __syncthreads();

  float sum = 0.f;
  for (int i = tid; i < n; i += 256) {
    float e = __expf(s[i] - m);
    sum += e;
    p[i] = (f16)e;
  }
  red[tid] = sum; __syncthreads();
  for (int off = 128; off > 0; off >>= 1) {
    if (tid < off) red[tid] += red[tid + off];
    __syncthreads();
  }
  const float inv = 1.0f / red[0];
  for (int i = tid; i < n; i += 256) p[i] = (f16)((float)p[i] * inv);
}

// ---------------------------------------------------------------- launcher
extern "C" void kernel_launch(void* const* d_in, const int* in_sizes, int n_in,
                              void* d_out, int out_size, void* d_ws, size_t ws_size,
                              hipStream_t stream) {
  (void)in_sizes; (void)n_in; (void)out_size; (void)ws_size;
  constexpr int Bb = 2, T = 2048, E = 1024, H = 16, L = 1024, HD = 64;
  constexpr int M = Bb * T;   // 4096
  const float scale = 0.125f; // 1/sqrt(64)

  const float* x  = (const float*)d_in[0];
  const float* Wq = (const float*)d_in[1];
  const float* bq = (const float*)d_in[2];
  const float* Wk = (const float*)d_in[3];
  const float* bk = (const float*)d_in[4];
  const float* Wv = (const float*)d_in[5];
  const float* bv = (const float*)d_in[6];
  const float* Wl = (const float*)d_in[7];
  const float* bl = (const float*)d_in[8];
  const float* Wo = (const float*)d_in[9];
  const float* bo = (const float*)d_in[10];

  // ---- carve workspace
  char* p = (char*)d_ws;
  auto a16 = [&](size_t elems) {
    f16* r = (f16*)p; p += ((elems * sizeof(f16) + 255) & ~(size_t)255); return r;
  };
  auto a32 = [&](size_t elems) {
    float* r = (float*)p; p += ((elems * sizeof(float) + 255) & ~(size_t)255); return r;
  };
  f16* x16   = a16((size_t)M * E);
  f16* WqT   = a16((size_t)E * E);   // [N=E][K=E]
  f16* WkT   = a16((size_t)E * E);
  f16* WvT   = a16((size_t)E * E);
  f16* WlT   = a16((size_t)L * E);   // [N=L][K=E]
  f16* WoT   = a16((size_t)E * L);   // [N=E][K=L]
  f16* q16   = a16((size_t)M * E);
  f16* kt16  = a16((size_t)M * E);
  f16* vt16  = a16((size_t)M * E);
  f16* k16   = a16((size_t)M * L);
  f16* v16   = a16((size_t)M * L);
  f16* v16t  = a16((size_t)M * L);   // per-batch transposed V: [b][L][T]
  f16* att16 = a16((size_t)M * L);
  f16* Pbuf  = a16((size_t)T * T);
  float* S   = a32((size_t)T * T);

  cvt_f32_to_f16<<<(M * E + 255) / 256, 256, 0, stream>>>(x, x16, M * E);
  auto trf = [&](const float* src, f16* dst, int R, int C) {
    transpose_to_f16<float><<<dim3(C / 32, R / 32), 256, 0, stream>>>(src, dst, R, C);
  };
  trf(Wq, WqT, E, E);   // Wq [E][E] -> WqT [E][E]
  trf(Wk, WkT, E, E);
  trf(Wv, WvT, E, E);
  trf(Wl, WlT, E, L);   // [E][L] -> [L][E]
  trf(Wo, WoT, L, E);   // [L][E] -> [E][L]

  auto grid_of = [](int Mi, int Ni) { return dim3((Ni + 127) / 128, (Mi + 127) / 128); };

  // ---- projections: q = x@Wq+bq ; kt = x@Wk+bk ; vt = x@Wv+bv
  gemm_wmma_f16<true><<<grid_of(M, E), 256, 0, stream>>>(
      x16, E, WqT, E, q16, E, bq, M, E, E, 1.0f);
  gemm_wmma_f16<true><<<grid_of(M, E), 256, 0, stream>>>(
      x16, E, WkT, E, kt16, E, bk, M, E, E, 1.0f);
  gemm_wmma_f16<true><<<grid_of(M, E), 256, 0, stream>>>(
      x16, E, WvT, E, vt16, E, bv, M, E, E, 1.0f);
  // ---- latent: k = kt@Wl+bl ; v = vt@Wl+bl
  gemm_wmma_f16<true><<<grid_of(M, L), 256, 0, stream>>>(
      kt16, E, WlT, E, k16, L, bl, M, L, E, 1.0f);
  gemm_wmma_f16<true><<<grid_of(M, L), 256, 0, stream>>>(
      vt16, E, WlT, E, v16, L, bl, M, L, E, 1.0f);

  // ---- transpose V per batch: v16 [T][L] -> v16t [L][T]
  for (int b = 0; b < Bb; ++b)
    transpose_to_f16<f16><<<dim3(L / 32, T / 32), 256, 0, stream>>>(
        v16 + (size_t)b * T * L, v16t + (size_t)b * L * T, T, L);

  // ---- attention per (b, h); S/P buffers reused (stream-ordered)
  for (int bh = 0; bh < Bb * H; ++bh) {
    const int b = bh / H, h = bh % H;
    const f16* qh  = q16  + (size_t)b * T * E + h * HD;        // [T, HD], lda = E
    const f16* kh  = k16  + (size_t)b * T * L + h * HD;        // [T, HD] = S's B^T
    const f16* vht = v16t + (size_t)b * L * T + (size_t)h * HD * T; // [HD][T]
    f16*       oh  = att16 + (size_t)b * T * L + h * HD;       // [T, HD], ldc = L

    // S = scale * q @ k^T   (Bt = kh: [N=T][K=HD])
    gemm_wmma_f16<false><<<grid_of(T, T), 256, 0, stream>>>(
        qh, E, kh, L, S, T, nullptr, T, T, HD, scale);
    // P = softmax(S) rows
    softmax_rows<<<T, 256, 0, stream>>>(S, Pbuf, T);
    // O = P @ v   (Bt = vht: [N=HD][K=T])
    gemm_wmma_f16<true><<<grid_of(T, HD), 256, 0, stream>>>(
        Pbuf, T, vht, T, oh, L, nullptr, T, HD, T, 1.0f);
  }

  // ---- output projection: out = att @ Wo + bo (f32 to d_out)
  gemm_wmma_f16<false><<<grid_of(M, E), 256, 0, stream>>>(
      att16, L, WoT, L, (float*)d_out, E, bo, M, E, L, 1.0f);
}
